// GATAValueActivation_50208167690483
// MI455X (gfx1250) — matrix operand order repaired and checked by reading, not compiled, not run on hardware
//
#include <hip/hip_runtime.h>

// ---------------------------------------------------------------------------
// GATA value-activation fused kernel for gfx1250 (MI455X, wave32, WMMA).
// bf16 V_WMMA_F32_16X16X32_BF16 with f32 accumulate; HBM traffic stays fp32,
// tiles converted to bf16 on the fly. Staging/epilogue are 128-bit vectorized.
// ---------------------------------------------------------------------------

typedef __attribute__((ext_vector_type(16))) __bf16 v16bf;
typedef __attribute__((ext_vector_type(8)))  __bf16 v8bf;
typedef __attribute__((ext_vector_type(8)))  float  v8f;
typedef __attribute__((ext_vector_type(4)))  float  v4f;

#define CCH    128      // C == SPH == EC == 128
#define SS     7        // S
#define SC     896      // S*C
#define DFULL  15
#define DUSE   13       // only degree-rows 0..12 of Xp are used
#define DOUT   14       // output rows per edge: 1 + 3 + 5 + 5
#define BE     16       // edges per block (one WMMA M-tile)
#define NTHREADS 256
#define NWAVES 8

// bf16 weight workspace layout (element offsets into d_ws)
#define WRS_OFF 0
#define GMA_OFF (SC * CCH)                  // 114688
#define XJ_OFF  (2 * SC * CCH)              // 229376
#define WTOT    (2 * SC * CCH + CCH * CCH)  // 245760 elems (~480 KB bf16)

// Convert 8 consecutive f32 -> packed v8bf (two b128 loads, packed cvts).
__device__ __forceinline__ v8bf cvt8(const float* __restrict__ p) {
  v4f a = *(const v4f*)p;
  v4f b = *(const v4f*)(p + 4);
  v8bf r;
  r[0] = (__bf16)a[0]; r[1] = (__bf16)a[1]; r[2] = (__bf16)a[2]; r[3] = (__bf16)a[3];
  r[4] = (__bf16)b[0]; r[5] = (__bf16)b[1]; r[6] = (__bf16)b[2]; r[7] = (__bf16)b[3];
  return r;
}

// --- A fragment (16x32 bf16), ISA 7.12.2 layout --------------------------
// lane (m = lane&15, h = lane>>4): VGPR 0..3 hold K = 8h .. 8h+7 (pairs),
// VGPR 4..7 hold K = 16+8h .. 16+8h+7. Two 16-byte LDS loads per fragment.
__device__ __forceinline__ v16bf load_a_frag(const __bf16* base, int row, int h, int k0) {
  union { v16bf v; v8bf p[2]; } u;
  const __bf16* r = base + row * CCH + k0;
  u.p[0] = *(const v8bf*)(r + 8 * h);
  u.p[1] = *(const v8bf*)(r + 16 + 8 * h);
  return u.v;
}

// --- B fragment (32x16 bf16) ---------------------------------------------
// lane (n = lane&15, h = lane>>4): VGPR 0..3 hold K = 8h .. 8h+7,
// VGPR 4..7 hold K = 16+8h .. 16+8h+7 (same order as the A fragment).
__device__ __forceinline__ v16bf load_b_frag(const __bf16* rowbase, int h, int k0) {
  union { v16bf v; v8bf p[2]; } u;
  const __bf16* r = rowbase + k0 + 8 * h;
  u.p[0] = *(const v8bf*)(r);
  u.p[1] = *(const v8bf*)(r + 16);
  return u.v;
}

__device__ __forceinline__ float silu_f(float x) {
  return x / (1.0f + __expf(-x));
}

// ---------------------------------------------------------------------------
// Prologue: convert the three weight matrices fp32 -> bf16 into d_ws once.
// ---------------------------------------------------------------------------
__global__ __launch_bounds__(256)
void cvt_weights_kernel(const float* __restrict__ wrs,
                        const float* __restrict__ gma,
                        const float* __restrict__ xjw,
                        __bf16* __restrict__ out) {
  int stride = gridDim.x * blockDim.x * 8;
  for (int i = (blockIdx.x * blockDim.x + threadIdx.x) * 8; i < WTOT; i += stride) {
    const float* src;
    if (i < GMA_OFF)      src = wrs + i;
    else if (i < XJ_OFF)  src = gma + (i - GMA_OFF);
    else                  src = xjw + (i - XJ_OFF);
    *(v8bf*)(out + i) = cvt8(src);
  }
}

// ---------------------------------------------------------------------------
// Main fused kernel: one block = 16 edges, 8 wave32s.
// ---------------------------------------------------------------------------
__global__ __launch_bounds__(NTHREADS)
void gata_value_act_kernel(const float* __restrict__ attn,
                           const float* __restrict__ t_ij,
                           const float* __restrict__ h_j,
                           const float* __restrict__ X_j,
                           const float* __restrict__ rl_ij,
                           const float* __restrict__ wrs_b,
                           const float* __restrict__ gma_b,
                           const __bf16* __restrict__ wbf,
                           float* __restrict__ out) {
  __shared__ __bf16 lds_t[BE * CCH];          //  4 KB
  __shared__ __bf16 lds_h[BE * CCH];          //  4 KB
  __shared__ __bf16 lds_X[DUSE * BE * CCH];   // 52 KB
  __shared__ float  lds_o[BE * SC];           // 56 KB (combined o = attn + t_bias)

  const int tid  = threadIdx.x;
  const int e0   = blockIdx.x * BE;
  const int lane = tid & 31;
  const int wv   = tid >> 5;
  const int lm   = lane & 15;   // A-row / BCD-column within the 16x16 tile
  const int lh   = lane >> 4;   // lane-half selector

  // ---- Phase 1: stage activations into LDS as bf16, 8 elems/thread -------
  // BE*CCH = 2048 elems = 256 threads * 8; thread tid owns [tid*8, tid*8+8).
  {
    const int i = tid * 8;
    const int e = i >> 7, k = i & 127;
    const size_t gofs = (size_t)(e0 + e) * CCH + k;
    *(v8bf*)(lds_t + i) = cvt8(t_ij + gofs);
    *(v8bf*)(lds_h + i) = cvt8(h_j + gofs);
#pragma unroll
    for (int d = 0; d < DUSE; ++d) {
      *(v8bf*)(lds_X + d * BE * CCH + i) =
          cvt8(X_j + ((size_t)(e0 + e) * DFULL + d) * CCH + k);
    }
  }
  __syncthreads();

  // A fragments for t and h are shared by every N-tile job: cache in VGPRs.
  v16bf a_t[4], a_h[4];
#pragma unroll
  for (int kb = 0; kb < 4; ++kb) {
    a_t[kb] = load_a_frag(lds_t, lm, lh, kb * 32);
    a_h[kb] = load_a_frag(lds_h, lm, lh, kb * 32);
  }

  const __bf16* wrs = wbf + WRS_OFF;
  const __bf16* gma = wbf + GMA_OFF;
  const __bf16* xjw = wbf + XJ_OFF;

  // ---- Phase 2: 56 combo jobs: both GEMM tiles + silu-gate + attn add -----
  for (int job = wv; job < SC / 16; job += NWAVES) {
    const int n0 = job * 16;
    v8f acc1 = {0.f, 0.f, 0.f, 0.f, 0.f, 0.f, 0.f, 0.f};
    v8f acc2 = {0.f, 0.f, 0.f, 0.f, 0.f, 0.f, 0.f, 0.f};
    const __bf16* wrow = wrs + (size_t)(n0 + lm) * CCH;
    const __bf16* grow = gma + (size_t)(n0 + lm) * CCH;
#pragma unroll
    for (int kb = 0; kb < 4; ++kb) {
      v16bf b1 = load_b_frag(wrow, lh, kb * 32);
      v16bf b2 = load_b_frag(grow, lh, kb * 32);
      acc1 = __builtin_amdgcn_wmma_f32_16x16x32_bf16(false, a_t[kb], false, b1,
                                                     (short)0, acc1, false, false);
      acc2 = __builtin_amdgcn_wmma_f32_16x16x32_bf16(false, a_h[kb], false, b2,
                                                     (short)0, acc2, false, false);
    }
    const float brs = wrs_b[n0 + lm];
    const float bg  = gma_b[n0 + lm];
#pragma unroll
    for (int j = 0; j < 8; ++j) {
      int row = j + 8 * lh;                       // edge index inside the tile
      float tl = acc1[j] + brs;
      float g  = acc2[j] + bg;
      float ov = attn[(size_t)(e0 + row) * SC + n0 + lm] + tl * silu_f(g);
      lds_o[row * SC + n0 + lm] = ov;
    }
  }
  __syncthreads();

  // ---- Phase 3a: scalar degree output: out[:,0] = silu(o[:,0]), 8/thread --
  {
    const int i = tid * 8;
    const int e = i >> 7, k = i & 127;
    v4f a = *(const v4f*)(lds_o + e * SC + k);
    v4f b = *(const v4f*)(lds_o + e * SC + k + 4);
    v4f ra, rb;
#pragma unroll
    for (int j = 0; j < 4; ++j) { ra[j] = silu_f(a[j]); rb[j] = silu_f(b[j]); }
    float* dst = out + (size_t)(e0 + e) * DOUT * CCH + k;
    *(v4f*)dst = ra;
    *(v4f*)(dst + 4) = rb;
  }

  // ---- Phase 3b: Xp jobs with fused degree epilogue -----------------------
  // Wave wv owns channel tile n0 = wv*16 for all d=0..12, so the xj_w B
  // fragments are loop-invariant per wave: cache all 4 K-blocks in VGPRs.
  {
    const int n0 = wv * 16;
    const __bf16* xrow = xjw + (size_t)(n0 + lm) * CCH;
    v16bf bx[4];
#pragma unroll
    for (int kb = 0; kb < 4; ++kb) bx[kb] = load_b_frag(xrow, lh, kb * 32);

    for (int d = 0; d < DUSE; ++d) {
      const int l = (d < 3) ? 0 : (d < 8) ? 1 : 2;
      v8f acc = {0.f, 0.f, 0.f, 0.f, 0.f, 0.f, 0.f, 0.f};
#pragma unroll
      for (int kb = 0; kb < 4; ++kb) {
        v16bf a = load_a_frag(lds_X + d * BE * CCH, lm, lh, kb * 32);
        acc = __builtin_amdgcn_wmma_f32_16x16x32_bf16(false, a, false, bx[kb],
                                                      (short)0, acc, false, false);
      }
#pragma unroll
      for (int j = 0; j < 8; ++j) {
        int row = j + 8 * lh;
        float o1 = lds_o[row * SC + (1 + l) * CCH + n0 + lm];
        float o2 = lds_o[row * SC + (4 + l) * CCH + n0 + lm];
        float rl = rl_ij[(size_t)(e0 + row) * DFULL + d];
        out[(size_t)(e0 + row) * DOUT * CCH + (1 + d) * CCH + n0 + lm] =
            o1 * rl + o2 * acc[j];
      }
    }
  }
}

// ---------------------------------------------------------------------------
extern "C" void kernel_launch(void* const* d_in, const int* in_sizes, int n_in,
                              void* d_out, int out_size, void* d_ws, size_t ws_size,
                              hipStream_t stream) {
  const float* attn  = (const float*)d_in[0];  // (E, S*C)
  const float* t_ij  = (const float*)d_in[1];  // (E, EC)
  const float* h_j   = (const float*)d_in[2];  // (E, SPH)
  const float* X_j   = (const float*)d_in[3];  // (E, D_FULL, SPH)
  const float* rl    = (const float*)d_in[4];  // (E, D_FULL)
  const float* wrs_w = (const float*)d_in[5];  // (S*C, EC)
  const float* wrs_b = (const float*)d_in[6];  // (S*C,)
  const float* gma_w = (const float*)d_in[7];  // (S*C, SPH)
  const float* gma_b = (const float*)d_in[8];  // (S*C,)
  const float* xj_w  = (const float*)d_in[9];  // (C, SPH)

  float*  out = (float*)d_out;
  __bf16* wbf = (__bf16*)d_ws;                 // needs WTOT*2 = 491,520 bytes

  const int E = in_sizes[1] / CCH;             // 65536

  hipLaunchKernelGGL(cvt_weights_kernel, dim3(120), dim3(256), 0, stream,
                     wrs_w, gma_w, xj_w, wbf);
  hipLaunchKernelGGL(gata_value_act_kernel, dim3(E / BE), dim3(NTHREADS), 0, stream,
                     attn, t_ij, h_j, X_j, rl, wrs_b, gma_b, wbf, out);
}
